// GraphEncodeProcessDecode_19250043421467
// MI455X (gfx1250) — matrix operand order repaired and checked by reading, not compiled, not run on hardware
//
#include <hip/hip_runtime.h>
#include <math.h>

typedef __attribute__((ext_vector_type(16))) _Float16 v16h;
typedef __attribute__((ext_vector_type(8)))  float    v8f;

namespace {
constexpr int Nn = 20000;
constexpr int Ne = 160000;
constexpr int Ll = 4;
}

// ---------------- WMMA building blocks (wave32, 16x16x32 f16 -> f32) ----------------

// A fragment from an LDS row-major f16 tile [16][ldk], K-tile kt (32 wide).
// ISA 16-bit A 16x32 layout: lane l holds row m=l&15;
// element e -> K = ((e&8)<<1) + ((l>>4)<<3) + (e&7) within the 32-K tile.
__device__ __forceinline__ v16h frag_a_lds(const _Float16* As, int ldk, int kt, int lane) {
  const _Float16* row = As + (lane & 15) * ldk + kt * 32 + ((lane >> 4) << 3);
  v16h a;
#pragma unroll
  for (int e = 0; e < 16; ++e) a[e] = row[((e & 8) << 1) + (e & 7)];
  return a;
}

// One 16x16 output tile: C = A(16 x 32*nkt) @ B + bias.  B is pre-packed so each
// lane reads 16 contiguous halves: packed[((ntile*nkt + kt)*32 + lane)*16 + e].
__device__ __forceinline__ v8f gemm_tile(const _Float16* As, int ldk, int nkt,
                                         const _Float16* __restrict__ Pw,
                                         const float* __restrict__ bias,
                                         int ntile, int lane) {
  v8f c = {};
  for (int kt = 0; kt < nkt; ++kt) {
    v16h a = frag_a_lds(As, ldk, kt, lane);
    v16h b = *(const v16h*)(Pw + ((size_t)((ntile * nkt + kt) * 32 + lane) << 4));
    c = __builtin_amdgcn_wmma_f32_16x16x32_f16(false, a, false, b, (short)0, c,
                                               false, false);
  }
  float bv = bias[(ntile << 4) + (lane & 15)];
#pragma unroll
  for (int r = 0; r < 8; ++r) c[r] += bv;
  return c;
}

// D layout: VGPR r holds row m = r + 8*(lane>=16), col n = lane&15.
__device__ __forceinline__ void store_tile_relu(v8f c, _Float16* Bs, int ntile, int lane) {
  int n = (ntile << 4) + (lane & 15);
  int mb = (lane >> 4) << 3;
#pragma unroll
  for (int r = 0; r < 8; ++r) {
    float v = c[r] > 0.0f ? c[r] : 0.0f;
    Bs[(mb + r) * 128 + n] = (_Float16)v;
  }
}

// ---------------- weight pre-pack: f32 [K][128] -> f16 B-fragment order --------------
// B 16-bit layout per 32-K tile: element e -> K = ((e&8)<<1) + ((lane>>4)<<3) + (e&7);
// column n = ntile*16 + (lane&15).
__global__ void pack_weight_kernel(const float* __restrict__ W, _Float16* __restrict__ P,
                                   int Korig, int Kpad) {
  int idx = blockIdx.x * blockDim.x + threadIdx.x;
  int total = Kpad * 128;
  if (idx >= total) return;
  int nkt = Kpad >> 5;
  int e = idx & 15;
  int lane = (idx >> 4) & 31;
  int rest = idx >> 9;
  int kt = rest % nkt;
  int ntile = rest / nkt;
  int K = kt * 32 + ((e & 8) << 1) + ((lane >> 4) << 3) + (e & 7);
  int n = (ntile << 4) + (lane & 15);
  float v = (K < Korig) ? W[K * 128 + n] : 0.0f;
  P[idx] = (_Float16)v;
}

__global__ void zero_f32(float* __restrict__ p, int n) {
  int i = blockIdx.x * blockDim.x + threadIdx.x;
  if (i < n) p[i] = 0.0f;
}

// ---------------- fused 3-layer encoder MLP (din -> 128 -> 128 -> 128) --------------
__global__ __launch_bounds__(256) void encode_kernel(
    const float* __restrict__ xin, int din,
    const float* __restrict__ mean, const float* __restrict__ stdv,
    const _Float16* __restrict__ P0, const float* __restrict__ b0,
    const _Float16* __restrict__ P1, const float* __restrict__ b1,
    const _Float16* __restrict__ P2, const float* __restrict__ b2,
    _Float16* __restrict__ out, _Float16* __restrict__ out2) {
  __shared__ _Float16 As[16 * 32];
  __shared__ _Float16 H1[16 * 128];
  __shared__ _Float16 H2[16 * 128];
  int t = threadIdx.x;
  int rBase = blockIdx.x * 16;
  for (int idx = t; idx < 16 * 32; idx += 256) {
    int r = idx >> 5, c = idx & 31;
    float v = 0.0f;
    if (c < din) v = (xin[(size_t)(rBase + r) * din + c] - mean[c]) / stdv[c];
    As[idx] = (_Float16)v;
  }
  __syncthreads();
  int wave = t >> 5, lane = t & 31;
  v8f c1 = gemm_tile(As, 32, 1, P0, b0, wave, lane);
  store_tile_relu(c1, H1, wave, lane);
  __syncthreads();
  v8f c2 = gemm_tile(H1, 128, 4, P1, b1, wave, lane);
  store_tile_relu(c2, H2, wave, lane);
  __syncthreads();
  v8f c3 = gemm_tile(H2, 128, 4, P2, b2, wave, lane);
  int n = (wave << 4) + (lane & 15);
  int mb = (lane >> 4) << 3;
#pragma unroll
  for (int r = 0; r < 8; ++r) {
    _Float16 hv = (_Float16)c3[r];
    size_t o = (size_t)(rBase + mb + r) * 128 + n;
    out[o] = hv;
    if (out2) out2[o] = hv;
  }
}

// ---------------- fused edge MLP (384->128->128->128) + scatter-sum -----------------
__global__ __launch_bounds__(256) void edge_update_kernel(
    const int* __restrict__ edge_index, const _Float16* __restrict__ h,
    _Float16* __restrict__ he, float* __restrict__ agg,
    const _Float16* __restrict__ P0, const float* __restrict__ b0,
    const _Float16* __restrict__ P1, const float* __restrict__ b1,
    const _Float16* __restrict__ P2, const float* __restrict__ b2,
    int E) {
  __shared__ unsigned int As_u[16 * 192];  // 16 x 384 halves
  __shared__ _Float16 H1[16 * 128];
  __shared__ _Float16 H2[16 * 128];
  __shared__ int ssrc[16];
  __shared__ int sdst[16];
  _Float16* As = (_Float16*)As_u;
  int t = threadIdx.x;
  int eBase = blockIdx.x * 16;
  if (t < 16) ssrc[t] = edge_index[eBase + t];
  else if (t < 32) sdst[t - 16] = edge_index[E + eBase + (t - 16)];
  __syncthreads();
  const unsigned int* he_u = (const unsigned int*)he;
  const unsigned int* h_u  = (const unsigned int*)h;
  for (int idx = t; idx < 16 * 192; idx += 256) {
    int r = idx / 192, c = idx - r * 192;
    unsigned int v;
    if (c < 64)       v = he_u[(size_t)(eBase + r) * 64 + c];
    else if (c < 128) v = h_u[(size_t)ssrc[r] * 64 + (c - 64)];
    else              v = h_u[(size_t)sdst[r] * 64 + (c - 128)];
    As_u[idx] = v;
  }
  __builtin_prefetch(he + (size_t)(eBase + 16) * 128, 0, 3);
  __syncthreads();
  int wave = t >> 5, lane = t & 31;
  v8f c1 = gemm_tile(As, 384, 12, P0, b0, wave, lane);
  store_tile_relu(c1, H1, wave, lane);
  __syncthreads();
  v8f c2 = gemm_tile(H1, 128, 4, P1, b1, wave, lane);
  store_tile_relu(c2, H2, wave, lane);
  __syncthreads();
  v8f c3 = gemm_tile(H2, 128, 4, P2, b2, wave, lane);
  int n = (wave << 4) + (lane & 15);
  int mb = (lane >> 4) << 3;
#pragma unroll
  for (int r = 0; r < 8; ++r) {
    int m = mb + r;
    float v = c3[r];
    he[(size_t)(eBase + m) * 128 + n] = (_Float16)v;
    atomicAdd(&agg[(size_t)sdst[m] * 128 + n], v);
  }
}

// ---------------- fused node MLP (256->128->128->128), h updated in place -----------
__global__ __launch_bounds__(256) void node_update_kernel(
    const float* __restrict__ agg, _Float16* __restrict__ h,
    const _Float16* __restrict__ P0, const float* __restrict__ b0,
    const _Float16* __restrict__ P1, const float* __restrict__ b1,
    const _Float16* __restrict__ P2, const float* __restrict__ b2) {
  __shared__ unsigned int As_u[16 * 128];  // 16 x 256 halves
  __shared__ _Float16 H1[16 * 128];
  __shared__ _Float16 H2[16 * 128];
  _Float16* As = (_Float16*)As_u;
  int t = threadIdx.x;
  int rBase = blockIdx.x * 16;
  const unsigned int* h_u = (const unsigned int*)h;
  for (int idx = t; idx < 16 * 64; idx += 256) {
    int r = idx >> 6, c = idx & 63;
    As_u[r * 128 + c] = h_u[(size_t)(rBase + r) * 64 + c];
  }
  for (int idx = t; idx < 16 * 128; idx += 256) {
    int r = idx >> 7, c = idx & 127;
    As[r * 256 + 128 + c] = (_Float16)agg[(size_t)(rBase + r) * 128 + c];
  }
  __syncthreads();
  int wave = t >> 5, lane = t & 31;
  v8f c1 = gemm_tile(As, 256, 8, P0, b0, wave, lane);
  store_tile_relu(c1, H1, wave, lane);
  __syncthreads();
  v8f c2 = gemm_tile(H1, 128, 4, P1, b1, wave, lane);
  store_tile_relu(c2, H2, wave, lane);
  __syncthreads();
  v8f c3 = gemm_tile(H2, 128, 4, P2, b2, wave, lane);
  int n = (wave << 4) + (lane & 15);
  int mb = (lane >> 4) << 3;
#pragma unroll
  for (int r = 0; r < 8; ++r)
    h[(size_t)(rBase + mb + r) * 128 + n] = (_Float16)c3[r];
}

// ---------------- decoder: tanh(h+h0) -> 128 -> 128 -> 3, denormalize ---------------
__global__ __launch_bounds__(256) void decode_kernel(
    const _Float16* __restrict__ h, const _Float16* __restrict__ h0,
    const _Float16* __restrict__ P0, const float* __restrict__ b0,
    const _Float16* __restrict__ P1, const float* __restrict__ b1,
    const float* __restrict__ W2, const float* __restrict__ b2,
    const float* __restrict__ om, const float* __restrict__ os,
    float* __restrict__ out) {
  __shared__ _Float16 As[16 * 128];
  __shared__ _Float16 H1[16 * 128];
  __shared__ float H2f[16 * 128];
  int t = threadIdx.x;
  int rBase = blockIdx.x * 16;
  for (int idx = t; idx < 16 * 128; idx += 256) {
    int r = idx >> 7, c = idx & 127;
    size_t o = (size_t)(rBase + r) * 128 + c;
    As[idx] = (_Float16)tanhf((float)h[o] + (float)h0[o]);
  }
  __syncthreads();
  int wave = t >> 5, lane = t & 31;
  v8f c1 = gemm_tile(As, 128, 4, P0, b0, wave, lane);
  store_tile_relu(c1, H1, wave, lane);
  __syncthreads();
  v8f c2 = gemm_tile(H1, 128, 4, P1, b1, wave, lane);
  {
    int n = (wave << 4) + (lane & 15);
    int mb = (lane >> 4) << 3;
#pragma unroll
    for (int r = 0; r < 8; ++r) {
      float v = c2[r] > 0.0f ? c2[r] : 0.0f;
      H2f[(mb + r) * 128 + n] = v;
    }
  }
  __syncthreads();
  if (t < 48) {
    int r = t / 3, j = t - r * 3;
    float acc = b2[j];
    for (int k = 0; k < 128; ++k) acc += H2f[r * 128 + k] * W2[k * 3 + j];
    out[(size_t)(rBase + r) * 3 + j] = acc * os[j] + om[j];
  }
}

// ------------------------------------ launch ----------------------------------------
extern "C" void kernel_launch(void* const* d_in, const int* in_sizes, int n_in,
                              void* d_out, int out_size, void* d_ws, size_t ws_size,
                              hipStream_t stream) {
  (void)in_sizes; (void)n_in; (void)out_size; (void)ws_size;
  const float* x         = (const float*)d_in[0];
  const float* edge_attr = (const float*)d_in[1];
  const int*   edge_index= (const int*)d_in[2];
  const float* node_mean = (const float*)d_in[3];
  const float* node_std  = (const float*)d_in[4];
  const float* edge_mean = (const float*)d_in[5];
  const float* edge_std  = (const float*)d_in[6];
  const float* out_mean  = (const float*)d_in[7];
  const float* out_std   = (const float*)d_in[8];
  const float* enW0=(const float*)d_in[9];  const float* enb0=(const float*)d_in[10];
  const float* enW1=(const float*)d_in[11]; const float* enb1=(const float*)d_in[12];
  const float* enW2=(const float*)d_in[13]; const float* enb2=(const float*)d_in[14];
  const float* eeW0=(const float*)d_in[15]; const float* eeb0=(const float*)d_in[16];
  const float* eeW1=(const float*)d_in[17]; const float* eeb1=(const float*)d_in[18];
  const float* eeW2=(const float*)d_in[19]; const float* eeb2=(const float*)d_in[20];
  const float* geW0=(const float*)d_in[21]; const float* geb0=(const float*)d_in[22];
  const float* geW1=(const float*)d_in[23]; const float* geb1=(const float*)d_in[24];
  const float* geW2=(const float*)d_in[25]; const float* geb2=(const float*)d_in[26];
  const float* gnW0=(const float*)d_in[27]; const float* gnb0=(const float*)d_in[28];
  const float* gnW1=(const float*)d_in[29]; const float* gnb1=(const float*)d_in[30];
  const float* gnW2=(const float*)d_in[31]; const float* gnb2=(const float*)d_in[32];
  const float* deW0=(const float*)d_in[33]; const float* deb0=(const float*)d_in[34];
  const float* deW1=(const float*)d_in[35]; const float* deb1=(const float*)d_in[36];
  const float* deW2=(const float*)d_in[37]; const float* deb2=(const float*)d_in[38];

  size_t off = 0;
  auto alloc = [&](size_t bytes) -> char* {
    char* p = (char*)d_ws + off;
    off += (bytes + 255) & ~(size_t)255;
    return p;
  };
  _Float16* h   = (_Float16*)alloc((size_t)Nn * 128 * 2);
  _Float16* h0  = (_Float16*)alloc((size_t)Nn * 128 * 2);
  _Float16* he  = (_Float16*)alloc((size_t)Ne * 128 * 2);
  float*    agg = (float*)   alloc((size_t)Nn * 128 * 4);
  auto allocW = [&](size_t Kpad) { return (_Float16*)alloc(Kpad * 128 * 2); };
  _Float16* PenW0 = allocW(32);  _Float16* PenW1 = allocW(128); _Float16* PenW2 = allocW(128);
  _Float16* PeeW0 = allocW(32);  _Float16* PeeW1 = allocW(128); _Float16* PeeW2 = allocW(128);
  _Float16* PgeW0 = allocW(384 * Ll); _Float16* PgeW1 = allocW(128 * Ll); _Float16* PgeW2 = allocW(128 * Ll);
  _Float16* PgnW0 = allocW(256 * Ll); _Float16* PgnW1 = allocW(128 * Ll); _Float16* PgnW2 = allocW(128 * Ll);
  _Float16* PdeW0 = allocW(128); _Float16* PdeW1 = allocW(128);

  auto pack = [&](const float* W, _Float16* P, int Korig, int Kpad) {
    int total = Kpad * 128;
    pack_weight_kernel<<<(total + 255) / 256, 256, 0, stream>>>(W, P, Korig, Kpad);
  };
  pack(enW0, PenW0, 16, 32);
  pack(enW1, PenW1, 128, 128);
  pack(enW2, PenW2, 128, 128);
  pack(eeW0, PeeW0, 8, 32);
  pack(eeW1, PeeW1, 128, 128);
  pack(eeW2, PeeW2, 128, 128);
  for (int l = 0; l < Ll; ++l) {
    pack(geW0 + (size_t)l * 384 * 128, PgeW0 + (size_t)l * 384 * 128, 384, 384);
    pack(geW1 + (size_t)l * 128 * 128, PgeW1 + (size_t)l * 128 * 128, 128, 128);
    pack(geW2 + (size_t)l * 128 * 128, PgeW2 + (size_t)l * 128 * 128, 128, 128);
    pack(gnW0 + (size_t)l * 256 * 128, PgnW0 + (size_t)l * 256 * 128, 256, 256);
    pack(gnW1 + (size_t)l * 128 * 128, PgnW1 + (size_t)l * 128 * 128, 128, 128);
    pack(gnW2 + (size_t)l * 128 * 128, PgnW2 + (size_t)l * 128 * 128, 128, 128);
  }
  pack(deW0, PdeW0, 128, 128);
  pack(deW1, PdeW1, 128, 128);

  encode_kernel<<<Nn / 16, 256, 0, stream>>>(x, 16, node_mean, node_std,
      PenW0, enb0, PenW1, enb1, PenW2, enb2, h, h0);
  encode_kernel<<<Ne / 16, 256, 0, stream>>>(edge_attr, 8, edge_mean, edge_std,
      PeeW0, eeb0, PeeW1, eeb1, PeeW2, eeb2, he, nullptr);

  for (int l = 0; l < Ll; ++l) {
    zero_f32<<<(Nn * 128 + 1023) / 1024, 1024, 0, stream>>>(agg, Nn * 128);
    edge_update_kernel<<<Ne / 16, 256, 0, stream>>>(edge_index, h, he, agg,
        PgeW0 + (size_t)l * 384 * 128, geb0 + l * 128,
        PgeW1 + (size_t)l * 128 * 128, geb1 + l * 128,
        PgeW2 + (size_t)l * 128 * 128, geb2 + l * 128, Ne);
    node_update_kernel<<<Nn / 16, 256, 0, stream>>>(agg, h,
        PgnW0 + (size_t)l * 256 * 128, gnb0 + l * 128,
        PgnW1 + (size_t)l * 128 * 128, gnb1 + l * 128,
        PgnW2 + (size_t)l * 128 * 128, gnb2 + l * 128);
  }
  decode_kernel<<<Nn / 16, 256, 0, stream>>>(h, h0, PdeW0, deb0, PdeW1, deb1,
      deW2, deb2, out_mean, out_std, (float*)d_out);
}
